// AVGAE_39273180955242
// MI455X (gfx1250) — matrix operand order repaired and checked by reading, not compiled, not run on hardware
//
#include <hip/hip_runtime.h>
#include <hip/hip_bf16.h>

// ---------------- types ----------------
typedef __bf16 bf16;
typedef __attribute__((ext_vector_type(16))) __bf16 v16bf;
typedef __attribute__((ext_vector_type(8)))  __bf16 v8bf;
typedef __attribute__((ext_vector_type(8)))  float  v8f;
typedef __attribute__((ext_vector_type(4)))  float  f32x4;
typedef __attribute__((ext_vector_type(4)))  int    i32x4;

#define NNODE 4096
#define LRELU_ALPHA 0.2f

__device__ __forceinline__ v8f wmma_bf16(v16bf a, v16bf b, v8f c) {
  // D = A(16x32 bf16) * B(32x16 bf16) + C(16x16 f32)
  return __builtin_amdgcn_wmma_f32_16x16x32_bf16(false, a, false, b, (short)0, c,
                                                 false, false);
}

__device__ __forceinline__ v16bf cat16(v8bf lo, v8bf hi) {
  return __builtin_shufflevector(lo, hi, 0,1,2,3,4,5,6,7,8,9,10,11,12,13,14,15);
}

// ---------------- small prep kernels ----------------
__global__ void cvt_f32_bf16(const float* __restrict__ in, bf16* __restrict__ out, int n) {
  int i = blockIdx.x * 256 + threadIdx.x;
  if (i < n) out[i] = (bf16)in[i];
}

// W[K][N] (row-major f32) -> Wt[N][K] (row-major bf16)
__global__ void transpose_cvt(const float* __restrict__ in, bf16* __restrict__ out,
                              int K, int N) {
  int i = blockIdx.x * 256 + threadIdx.x;
  if (i < K * N) {
    int k = i / N, n = i % N;
    out[(size_t)n * K + k] = (bf16)in[i];
  }
}

// f1[i] = sum_n hT[n][i]*a[n] ; f2[i] = sum_n hT[n][i]*a[D+n]
__global__ void compute_f12(const bf16* __restrict__ hT, const float* __restrict__ a,
                            float* __restrict__ f1, float* __restrict__ f2, int D) {
  int i = blockIdx.x * 256 + threadIdx.x;  // i < NNODE, coalesced over i
  float s1 = 0.f, s2 = 0.f;
  for (int n = 0; n < D; ++n) {
    float hv = (float)hT[(size_t)n * NNODE + i];
    s1 += hv * a[n];
    s2 += hv * a[D + n];
  }
  f1[i] = s1;
  f2[i] = s2;
}

// m2[row] = max_{j : adj[row][j]>0} f2[j]   (rank-1 logits => row max is cheap)
__global__ void row_masked_max(const int* __restrict__ adj, const float* __restrict__ f2,
                               float* __restrict__ m2) {
  __shared__ float red[256];
  int row = blockIdx.x;
  const int* ar = adj + (size_t)row * NNODE;
  float mx = -3.0e38f;
  for (int j = threadIdx.x; j < NNODE; j += 256)
    if (ar[j] > 0) mx = fmaxf(mx, f2[j]);
  red[threadIdx.x] = mx;
  __syncthreads();
  for (int s = 128; s > 0; s >>= 1) {
    if (threadIdx.x < s) red[threadIdx.x] = fmaxf(red[threadIdx.x], red[threadIdx.x + s]);
    __syncthreads();
  }
  if (threadIdx.x == 0) m2[row] = red[0];
}

// ---------------- GEMM: Ct[N][M] (bf16, transposed) = (A[M][K] @ Bt[N][K]^T)^T ----------
// per wave: 16 rows x 64 cols output, K-loop step 32, 4 WMMAs/step
__global__ __launch_bounds__(256) void gemm_ht(const bf16* __restrict__ A,
                                               const bf16* __restrict__ Bt,
                                               bf16* __restrict__ Ct,
                                               int M, int N, int K) {
  const int lane   = threadIdx.x & 31;
  const int wid    = threadIdx.x >> 5;
  const int lane16 = lane & 15;
  const bool hi    = lane >= 16;
  const int hi8    = hi ? 8 : 0;
  const int nT     = N >> 6;
  const int total  = (M >> 4) * nT;
  int tile = blockIdx.x * 8 + wid;
  if (tile >= total) return;
  const int rowTile = tile / nT;
  const int nTile   = tile % nT;

  v8f acc[4] = {};
  const bf16* aRow  = A  + (size_t)(rowTile * 16 + lane16) * K + (hi ? 16 : 0);
  const bf16* bBase = Bt + (size_t)(nTile * 64 + lane16) * K + (hi ? 16 : 0);

  for (int k0 = 0; k0 < K; k0 += 32) {
    v8bf alo = *(const v8bf*)(aRow + k0);
    v8bf ahi = *(const v8bf*)(aRow + k0 + 8);
    v16bf a  = cat16(alo, ahi);
#pragma unroll
    for (int nt = 0; nt < 4; ++nt) {
      v16bf b = *(const v16bf*)(bBase + (size_t)nt * 16 * K + k0);
      acc[nt] = wmma_bf16(a, b, acc[nt]);
    }
  }
  // C layout: lane16 = col, elem v = row v (+8 for hi lanes). Store transposed bf16.
#pragma unroll
  for (int nt = 0; nt < 4; ++nt) {
    v8bf o;
#pragma unroll
    for (int v = 0; v < 8; ++v) o[v] = (bf16)acc[nt][v];
    *(v8bf*)(Ct + (size_t)(nTile * 64 + nt * 16 + lane16) * M + rowTile * 16 + hi8) = o;
  }
}

// ---------------- layer-0 attention (D=256), LDS-cooperative ----------------
// One 16-row tile per 128-thread block (4 waves). P fragments are a pure function of
// lane id, so waves produce them round-robin into LDS (exp computed ONCE per element,
// adj row read ONCE) and every wave consumes them for its own 64-col d-slice.
__global__ __launch_bounds__(128) void gat_attn0_lds(const int* __restrict__ adj,
                                                     const float* __restrict__ f1,
                                                     const float* __restrict__ f2,
                                                     const float* __restrict__ m2,
                                                     const bf16* __restrict__ hT,
                                                     bf16* __restrict__ outB) {
  constexpr int D  = 256;
  constexpr int CH = 32;  // j-steps (32 cols each) per LDS chunk
  __shared__ __align__(32) bf16 Plds[CH * 32 * 16];  // 32 KiB of P fragments
  __shared__ float sden[4][16];

  const int lane   = threadIdx.x & 31;
  const int wid    = threadIdx.x >> 5;  // 0..3
  const int lane16 = lane & 15;
  const bool hi    = lane >= 16;
  const int hiA    = hi ? 8 : 0;
  const int hiB    = hi ? 16 : 0;
  const int rowTile = blockIdx.x;  // 256 blocks
  const int row     = rowTile * 16 + lane16;

  const float f1r = f1[row];
  float t0 = f1r + m2[row];
  const float mrow = t0 > 0.f ? t0 : LRELU_ALPHA * t0;
  const int* adjRow = adj + (size_t)row * NNODE;

  v8f acc[4] = {};
  float dsum = 0.f;

  auto pelem = [&](int av, float f2v) -> bf16 {
    float t = f1r + f2v;
    t = t > 0.f ? t : LRELU_ALPHA * t;
    float pv = (av > 0) ? __expf(t - mrow) : 0.f;
    dsum += pv;
    return (bf16)pv;
  };

  const bf16* hTb = hT + (size_t)(wid * 64 + lane16) * NNODE + hiB;

  for (int c = 0; c < NNODE / 32; c += CH) {
    // ---- phase 1: produce P fragments for this chunk (round robin over waves)
    for (int s = c + wid; s < c + CH; s += 4) {
      const int j0 = s * 32;
      const int jA = j0 + hiB;
      const int jB = j0 + hiB + 8;
      i32x4 ad0 = *(const i32x4*)(adjRow + jA);
      i32x4 ad1 = *(const i32x4*)(adjRow + jA + 4);
      i32x4 ad2 = *(const i32x4*)(adjRow + jB);
      i32x4 ad3 = *(const i32x4*)(adjRow + jB + 4);
      f32x4 f20 = *(const f32x4*)(f2 + jA);
      f32x4 f21 = *(const f32x4*)(f2 + jA + 4);
      f32x4 f22 = *(const f32x4*)(f2 + jB);
      f32x4 f23 = *(const f32x4*)(f2 + jB + 4);
      v16bf p;
#pragma unroll
      for (int e = 0; e < 4; ++e) {
        p[e]      = pelem(ad0[e], f20[e]);
        p[4 + e]  = pelem(ad1[e], f21[e]);
        p[8 + e]  = pelem(ad2[e], f22[e]);
        p[12 + e] = pelem(ad3[e], f23[e]);
      }
      *(v16bf*)&Plds[((s - c) * 32 + lane) * 16] = p;
    }
    __syncthreads();
    // ---- phase 2: every wave consumes the chunk for its 64-col d-slice
    for (int s = c; s < c + CH; ++s) {
      v16bf p = *(const v16bf*)&Plds[((s - c) * 32 + lane) * 16];
      const bf16* hTp = hTb + s * 32;
#pragma unroll
      for (int nt = 0; nt < 4; ++nt) {
        v16bf b = *(const v16bf*)(hTp + (size_t)nt * 16 * NNODE);
        acc[nt] = wmma_bf16(p, b, acc[nt]);
      }
    }
    __syncthreads();
  }

  // denominator: lanes m / m+16 cover disjoint column subsets; waves cover disjoint j
  float dall = dsum + __shfl_xor(dsum, 16, 32);
  if (!hi) sden[wid][lane16] = dall;
  __syncthreads();
  float scale[8];
#pragma unroll
  for (int v = 0; v < 8; ++v) {
    int r = v + hiA;
    float dv = sden[0][r] + sden[1][r] + sden[2][r] + sden[3][r];
    scale[v] = 1.0f / (dv + 1e-30f);
  }
#pragma unroll
  for (int nt = 0; nt < 4; ++nt) {
    const int col = wid * 64 + nt * 16 + lane16;
#pragma unroll
    for (int v = 0; v < 8; ++v) {
      const int r = rowTile * 16 + v + hiA;
      outB[(size_t)r * D + col] = (bf16)(acc[nt][v] * scale[v]);
    }
  }
}

// ---------------- attention for D=64 (layers 1/2): one wave per 16-row tile --------
template <int D>
__global__ __launch_bounds__(256) void gat_attn(const int* __restrict__ adj,
                                                const float* __restrict__ f1,
                                                const float* __restrict__ f2,
                                                const float* __restrict__ m2,
                                                const bf16* __restrict__ hT,
                                                float* __restrict__ outF) {
  const int lane   = threadIdx.x & 31;
  const int wid    = threadIdx.x >> 5;
  const int lane16 = lane & 15;
  const bool hi    = lane >= 16;
  const int hiA    = hi ? 8 : 0;
  const int hiB    = hi ? 16 : 0;
  const int total  = NNODE / 16;
  int tile = blockIdx.x * 8 + wid;
  if (tile >= total) return;
  const int rowTile = tile;

  const int row = rowTile * 16 + lane16;
  const float f1r = f1[row];
  float t0 = f1r + m2[row];
  const float mrow = t0 > 0.f ? t0 : LRELU_ALPHA * t0;
  const int* adjRow = adj + (size_t)row * NNODE;

  v8f acc[4] = {};
  float dsum = 0.f;

  auto pelem = [&](int av, float f2v) -> bf16 {
    float t = f1r + f2v;
    t = t > 0.f ? t : LRELU_ALPHA * t;
    float pv = (av > 0) ? __expf(t - mrow) : 0.f;
    dsum += pv;
    return (bf16)pv;
  };

  for (int j0 = 0; j0 < NNODE; j0 += 32) {
    const int jA = j0 + hiB;
    const int jB = j0 + hiB + 8;
    i32x4 ad0 = *(const i32x4*)(adjRow + jA);
    i32x4 ad1 = *(const i32x4*)(adjRow + jA + 4);
    i32x4 ad2 = *(const i32x4*)(adjRow + jB);
    i32x4 ad3 = *(const i32x4*)(adjRow + jB + 4);
    f32x4 f20 = *(const f32x4*)(f2 + jA);
    f32x4 f21 = *(const f32x4*)(f2 + jA + 4);
    f32x4 f22 = *(const f32x4*)(f2 + jB);
    f32x4 f23 = *(const f32x4*)(f2 + jB + 4);

    v16bf p;
#pragma unroll
    for (int e = 0; e < 4; ++e) {
      p[e]      = pelem(ad0[e], f20[e]);
      p[4 + e]  = pelem(ad1[e], f21[e]);
      p[8 + e]  = pelem(ad2[e], f22[e]);
      p[12 + e] = pelem(ad3[e], f23[e]);
    }

    const bf16* hTp = hT + (size_t)lane16 * NNODE + j0 + hiB;
#pragma unroll
    for (int nt = 0; nt < 4; ++nt) {
      v16bf b = *(const v16bf*)(hTp + (size_t)nt * 16 * NNODE);
      acc[nt] = wmma_bf16(p, b, acc[nt]);
    }
  }

  float dall = dsum + __shfl_xor(dsum, 16, 32);
  float scale[8];
#pragma unroll
  for (int v = 0; v < 8; ++v) {
    float dv = __shfl(dall, v + hiA, 32);
    scale[v] = 1.0f / (dv + 1e-30f);
  }

#pragma unroll
  for (int nt = 0; nt < 4; ++nt) {
    const int col = nt * 16 + lane16;
#pragma unroll
    for (int v = 0; v < 8; ++v) {
      const int r = rowTile * 16 + v + hiA;
      outF[(size_t)r * D + col] = acc[nt][v] * scale[v];
    }
  }
}

// ---------------- Z = noise*exp(logstd)+mean  (bf16 row-major) ----------------
__global__ void make_z(const float* __restrict__ noise, const float* __restrict__ mean,
                       const float* __restrict__ logstd, bf16* __restrict__ Zb, int n) {
  int i = blockIdx.x * 256 + threadIdx.x;
  if (i < n) Zb[i] = (bf16)(noise[i] * __expf(logstd[i]) + mean[i]);
}

// ---------------- A_pred = sigmoid(Z @ Z^T), streamed out non-temporally -------------
__global__ __launch_bounds__(256) void zzt_sigmoid(const bf16* __restrict__ Zb,
                                                   float* __restrict__ out) {
  const int lane   = threadIdx.x & 31;
  const int wid    = threadIdx.x >> 5;
  const int lane16 = lane & 15;
  const bool hi    = lane >= 16;
  const int hiA    = hi ? 8 : 0;
  const int total  = (NNODE / 16) * (NNODE / 64);
  int tile = blockIdx.x * 8 + wid;
  if (tile >= total) return;
  const int rowTile = tile >> 6;  // NNODE/64 = 64 col tiles
  const int cTile   = tile & 63;

  v8f acc[4] = {};
  const bf16* aRow = Zb + (size_t)(rowTile * 16 + lane16) * 64 + (hi ? 16 : 0);
  const bf16* bRow = Zb + (size_t)(cTile * 64 + lane16) * 64 + (hi ? 16 : 0);
#pragma unroll
  for (int k0 = 0; k0 < 64; k0 += 32) {
    v8bf alo = *(const v8bf*)(aRow + k0);
    v8bf ahi = *(const v8bf*)(aRow + k0 + 8);
    v16bf a  = cat16(alo, ahi);
#pragma unroll
    for (int nt = 0; nt < 4; ++nt) {
      v16bf b = *(const v16bf*)(bRow + (size_t)nt * 16 * 64 + k0);
      acc[nt] = wmma_bf16(a, b, acc[nt]);
    }
  }
#pragma unroll
  for (int nt = 0; nt < 4; ++nt) {
    const int col = cTile * 64 + nt * 16 + lane16;
#pragma unroll
    for (int v = 0; v < 8; ++v) {
      const int r = rowTile * 16 + v + hiA;
      float s = 1.0f / (1.0f + __expf(-acc[nt][v]));
      // NT store: 64 MiB stream-out, keep adj/hT resident in the 192MB L2
      __builtin_nontemporal_store(s, &out[(size_t)r * NNODE + col]);
    }
  }
}

// ---------------- host-side orchestration ----------------
extern "C" void kernel_launch(void* const* d_in, const int* in_sizes, int n_in,
                              void* d_out, int out_size, void* d_ws, size_t ws_size,
                              hipStream_t stream) {
  const float* X     = (const float*)d_in[0];  // [4096,512]
  const int*   adj   = (const int*)  d_in[1];  // [4096,4096]
  const float* noise = (const float*)d_in[2];  // [4096,64]
  const float* W0    = (const float*)d_in[3];  // [512,256]
  const float* a0    = (const float*)d_in[4];  // [512]
  const float* W1    = (const float*)d_in[5];  // [256,64]
  const float* a1    = (const float*)d_in[6];  // [128]
  const float* W2    = (const float*)d_in[7];  // [256,64]
  const float* a2    = (const float*)d_in[8];  // [128]
  float* out = (float*)d_out;                  // [4096,4096]

  // workspace bump allocator (~12.5 MiB total)
  char* w = (char*)d_ws;
  auto alloc = [&](size_t bytes) -> void* {
    void* p = (void*)w;
    w += (bytes + 255) & ~(size_t)255;
    return p;
  };
  bf16* Xb      = (bf16*)alloc((size_t)4096 * 512 * 2);
  bf16* W0t     = (bf16*)alloc((size_t)256 * 512 * 2);
  bf16* W1t     = (bf16*)alloc((size_t)64 * 256 * 2);
  bf16* W2t     = (bf16*)alloc((size_t)64 * 256 * 2);
  bf16* h0T     = (bf16*)alloc((size_t)256 * 4096 * 2);
  bf16* hiddenB = (bf16*)alloc((size_t)4096 * 256 * 2);
  bf16* h1T     = (bf16*)alloc((size_t)64 * 4096 * 2);
  bf16* h2T     = (bf16*)alloc((size_t)64 * 4096 * 2);
  float* meanF  = (float*)alloc((size_t)4096 * 64 * 4);
  float* lstdF  = (float*)alloc((size_t)4096 * 64 * 4);
  bf16* Zb      = (bf16*)alloc((size_t)4096 * 64 * 2);
  float *f1v[3], *f2v[3], *m2v[3];
  for (int l = 0; l < 3; ++l) {
    f1v[l] = (float*)alloc(4096 * 4);
    f2v[l] = (float*)alloc(4096 * 4);
    m2v[l] = (float*)alloc(4096 * 4);
  }

  // prep: bf16 conversions / weight transposes
  cvt_f32_bf16<<<(4096 * 512 + 255) / 256, 256, 0, stream>>>(X, Xb, 4096 * 512);
  transpose_cvt<<<(512 * 256 + 255) / 256, 256, 0, stream>>>(W0, W0t, 512, 256);
  transpose_cvt<<<(256 * 64 + 255) / 256, 256, 0, stream>>>(W1, W1t, 256, 64);
  transpose_cvt<<<(256 * 64 + 255) / 256, 256, 0, stream>>>(W2, W2t, 256, 64);

  // layer 0: h0 = X@W0 (stored transposed bf16), f1/f2, masked rowmax, attention
  gemm_ht<<<128, 256, 0, stream>>>(Xb, W0t, h0T, 4096, 256, 512);  // 1024 tiles
  compute_f12<<<16, 256, 0, stream>>>(h0T, a0, f1v[0], f2v[0], 256);
  row_masked_max<<<4096, 256, 0, stream>>>(adj, f2v[0], m2v[0]);
  gat_attn0_lds<<<256, 128, 0, stream>>>(adj, f1v[0], f2v[0], m2v[0], h0T, hiddenB);

  // layer 1 (mean) and layer 2 (logstd)
  gemm_ht<<<32, 256, 0, stream>>>(hiddenB, W1t, h1T, 4096, 64, 256);  // 256 tiles
  compute_f12<<<16, 256, 0, stream>>>(h1T, a1, f1v[1], f2v[1], 64);
  row_masked_max<<<4096, 256, 0, stream>>>(adj, f2v[1], m2v[1]);
  gat_attn<64><<<32, 256, 0, stream>>>(adj, f1v[1], f2v[1], m2v[1], h1T, meanF);

  gemm_ht<<<32, 256, 0, stream>>>(hiddenB, W2t, h2T, 4096, 64, 256);
  compute_f12<<<16, 256, 0, stream>>>(h2T, a2, f1v[2], f2v[2], 64);
  row_masked_max<<<4096, 256, 0, stream>>>(adj, f2v[2], m2v[2]);
  gat_attn<64><<<32, 256, 0, stream>>>(adj, f1v[2], f2v[2], m2v[2], h2T, lstdF);

  // Z and decoder
  make_z<<<(4096 * 64 + 255) / 256, 256, 0, stream>>>(noise, meanF, lstdF, Zb, 4096 * 64);
  zzt_sigmoid<<<2048, 256, 0, stream>>>(Zb, out);  // 16384 tiles
}
